// AttentivePool_49057116455034
// MI455X (gfx1250) — matrix operand order repaired and checked by reading, not compiled, not run on hardware
//
#include <hip/hip_runtime.h>
#include <hip/hip_bf16.h>
#include <stdint.h>

#define B_ 16
#define C_ 128
#define H_ 32
#define W_ 4096
#define M_ (W_ * B_)   // 65536 rows = (w,b) pairs

typedef __attribute__((ext_vector_type(16))) __bf16 v16bf;
typedef __attribute__((ext_vector_type(8)))  float  v8f;

__device__ __forceinline__ uint16_t f32_bf16(float f) {
  union { float f; uint32_t u; } c; c.f = f;
  uint32_t u = c.u;
  return (uint16_t)((u + 0x7FFFu + ((u >> 16) & 1u)) >> 16);  // round-to-nearest-even
}
__device__ __forceinline__ float bf16_f32(uint16_t h) {
  union { uint32_t u; float f; } c; c.u = ((uint32_t)h) << 16;
  return c.f;
}
__device__ __forceinline__ uint32_t pack2(float lo, float hi) {
  return (uint32_t)f32_bf16(lo) | ((uint32_t)f32_bf16(hi) << 16);
}

// ---------------------------------------------------------------------------
// Kernel 1: mean over H, transpose to A-matrix layout [M = w*B+b][K = c], bf16.
// Coalesced reads of x (64 consecutive w per row segment), LDS transpose,
// 32B-granular writes of bf16 rows.
// ---------------------------------------------------------------------------
__global__ __launch_bounds__(256)
void pool_transpose_kernel(const float* __restrict__ x, uint16_t* __restrict__ poolT) {
  __shared__ float tile[16][65];          // [c_local][w_local], padded
  const int tid = threadIdx.x;
  const int b  = blockIdx.z;
  const int c0 = blockIdx.y * 16;
  const int w0 = blockIdx.x * 64;
  const int wl = tid & 63;
  const int cb = tid >> 6;                // 0..3
  #pragma unroll
  for (int it = 0; it < 4; ++it) {
    const int cl = it * 4 + cb;           // 0..15
    const float* p = x + ((size_t)(b * C_ + c0 + cl)) * (size_t)(H_ * W_) + (w0 + wl);
    float s = 0.f;
    #pragma unroll
    for (int h = 0; h < H_; ++h) s += p[(size_t)h * W_];
    tile[cl][wl] = s * (1.0f / H_);
  }
  __syncthreads();
  const int wlo = tid >> 2;               // 0..63
  const int cg  = (tid & 3) * 4;          // 0,4,8,12
  const int m   = (w0 + wlo) * B_ + b;
  uint32_t d0 = pack2(tile[cg + 0][wlo], tile[cg + 1][wlo]);
  uint32_t d1 = pack2(tile[cg + 2][wlo], tile[cg + 3][wlo]);
  *(uint2*)(poolT + (size_t)m * C_ + c0 + cg) = make_uint2(d0, d1);
}

// ---------------------------------------------------------------------------
// Kernel 2: attn[m] = sum_n proj[n] * tanh( sum_k A[m][k]*W[k][n] + bias[n] )
// via v_wmma_f32_16x16x32_bf16. Weight packed into LDS as bf16 K-pairs in the
// B-fragment layout; each wave does 8 N-tiles x 4 K-steps = 32 WMMAs.
// ---------------------------------------------------------------------------
__global__ __launch_bounds__(256)
void attn_gemm_kernel(const uint16_t* __restrict__ poolT,
                      const float* __restrict__ Wg,
                      const float* __restrict__ proj,
                      const float* __restrict__ bias,
                      float* __restrict__ attn) {
  __shared__ uint32_t wlds[64 * 128];     // [kpair][n] = {W[2p][n], W[2p+1][n]} bf16x2, 32 KB
  __shared__ float bias_s[128];
  __shared__ float proj_s[128];
  const int tid = threadIdx.x;
  for (int i = tid; i < 64 * 128; i += 256) {
    const int p = i >> 7;
    const int n = i & 127;
    wlds[i] = pack2(Wg[(size_t)(2 * p) * C_ + n], Wg[(size_t)(2 * p + 1) * C_ + n]);
  }
  if (tid < 128) { bias_s[tid] = bias[tid]; proj_s[tid] = proj[tid]; }
  __syncthreads();

  const int wave = tid >> 5;
  const int lane = tid & 31;
  const int h    = lane >> 4;             // half-wave
  const int ln   = lane & 15;
  const int r0   = blockIdx.x * 128 + wave * 16;

  // A fragments: this code's contiguous k order, same as B. Lane row = ln; VGPR v holds K pair:
  //   K = 2v + 16h
  const uint16_t* arow = poolT + (size_t)(r0 + ln) * C_;
  uint32_t afrag[4][8];
  #pragma unroll
  for (int kt = 0; kt < 4; ++kt) {
    #pragma unroll
    for (int v = 0; v < 8; ++v) {
      const int kb = kt * 32 + 2 * v + 16 * h;
      afrag[kt][v] = *(const uint32_t*)(arow + kb);
    }
  }

  float rowacc[8];
  #pragma unroll
  for (int v = 0; v < 8; ++v) rowacc[v] = 0.f;

  union U { v16bf v; uint32_t u[8]; };

  #pragma unroll
  for (int nt = 0; nt < 8; ++nt) {
    const int n = nt * 16 + ln;
    v8f c = {0.f, 0.f, 0.f, 0.f, 0.f, 0.f, 0.f, 0.f};
    #pragma unroll
    for (int kt = 0; kt < 4; ++kt) {
      U ua, ub;
      #pragma unroll
      for (int v = 0; v < 8; ++v) {
        ua.u[v] = afrag[kt][v];
        // B fragment: VGPR v, lanes 0-15: K=2v,2v+1 ; lanes 16-31: K=16+2v,...
        ub.u[v] = wlds[(kt * 16 + v + 8 * h) * 128 + n];
      }
      c = __builtin_amdgcn_wmma_f32_16x16x32_bf16(
              /*neg_a=*/false, ua.v, /*neg_b=*/false, ub.v,
              /*c_mod=*/(short)0, c, /*reuse_a=*/false, /*reuse_b=*/false);
    }
    const float bi = bias_s[n];
    const float pj = proj_s[n];
    #pragma unroll
    for (int v = 0; v < 8; ++v) rowacc[v] += tanhf(c[v] + bi) * pj;
  }
  // Reduce over N (16 lanes of each half; C-layout row = r0 + v + 8h)
  #pragma unroll
  for (int mask = 1; mask <= 8; mask <<= 1) {
    #pragma unroll
    for (int v = 0; v < 8; ++v) rowacc[v] += __shfl_xor(rowacc[v], mask, 32);
  }
  if (ln == 0) {
    #pragma unroll
    for (int v = 0; v < 8; ++v) attn[r0 + v + 8 * h] = rowacc[v];
  }
}

// ---------------------------------------------------------------------------
// Kernel 3: softmax over W (4096) per batch b. One block per b.
// ---------------------------------------------------------------------------
__global__ __launch_bounds__(256)
void softmax_kernel(const float* __restrict__ attn, float* __restrict__ attn_norm) {
  const int b = blockIdx.x;
  const int tid = threadIdx.x;
  const int lane = tid & 31, wave = tid >> 5;
  __shared__ float redmax[8];
  __shared__ float redsum[8];
  float vals[16];
  #pragma unroll
  for (int i = 0; i < 16; ++i) vals[i] = attn[(size_t)(tid + i * 256) * B_ + b];
  float m = vals[0];
  #pragma unroll
  for (int i = 1; i < 16; ++i) m = fmaxf(m, vals[i]);
  #pragma unroll
  for (int mask = 1; mask < 32; mask <<= 1) m = fmaxf(m, __shfl_xor(m, mask, 32));
  if (lane == 0) redmax[wave] = m;
  __syncthreads();
  float bm = redmax[0];
  #pragma unroll
  for (int j = 1; j < 8; ++j) bm = fmaxf(bm, redmax[j]);
  float s = 0.f;
  #pragma unroll
  for (int i = 0; i < 16; ++i) { vals[i] = __expf(vals[i] - bm); s += vals[i]; }
  #pragma unroll
  for (int mask = 1; mask < 32; mask <<= 1) s += __shfl_xor(s, mask, 32);
  if (lane == 0) redsum[wave] = s;
  __syncthreads();
  float bs = 0.f;
  #pragma unroll
  for (int j = 0; j < 8; ++j) bs += redsum[j];
  const float inv = 1.0f / bs;
  #pragma unroll
  for (int i = 0; i < 16; ++i)
    attn_norm[(size_t)(tid + i * 256) * B_ + b] = vals[i] * inv;
}

// ---------------------------------------------------------------------------
// Kernel 4: partial weighted sums over w-chunks (deterministic, no atomics).
// out[b][c] = sum_w attn_norm[w,b] * pooled[w,b,c]; pooled rows are 256B,
// read coalesced by the 128 c-threads.
// ---------------------------------------------------------------------------
__global__ __launch_bounds__(128)
void wsum_partial_kernel(const uint16_t* __restrict__ poolT,
                         const float* __restrict__ attn_norm,
                         float* __restrict__ partial) {
  const int c = threadIdx.x;
  const int chunk = blockIdx.x;           // 16 chunks of 256 w
  const int b = blockIdx.y;
  float acc = 0.f;
  for (int wi = 0; wi < 256; ++wi) {
    const int m = (chunk * 256 + wi) * B_ + b;
    const float a = attn_norm[m];
    acc += a * bf16_f32(poolT[(size_t)m * C_ + c]);
  }
  partial[(size_t)(chunk * B_ + b) * C_ + c] = acc;
}

__global__ __launch_bounds__(256)
void wsum_reduce_kernel(const float* __restrict__ partial, float* __restrict__ out) {
  const int idx = blockIdx.x * 256 + threadIdx.x;   // idx = b*C + c, 0..2047
  float acc = 0.f;
  #pragma unroll
  for (int ch = 0; ch < 16; ++ch) acc += partial[(size_t)ch * (B_ * C_) + idx];
  out[idx] = acc;
}

// ---------------------------------------------------------------------------
extern "C" void kernel_launch(void* const* d_in, const int* in_sizes, int n_in,
                              void* d_out, int out_size, void* d_ws, size_t ws_size,
                              hipStream_t stream) {
  const float* x    = (const float*)d_in[0];   // [16,128,32,4096]
  const float* Wg   = (const float*)d_in[1];   // [128,128]
  const float* proj = (const float*)d_in[2];   // [128,1]
  const float* bias = (const float*)d_in[3];   // [128,1]
  float* out = (float*)d_out;                  // [16,128]

  char* ws = (char*)d_ws;
  uint16_t* poolT  = (uint16_t*)ws;                          // 16 MB bf16 A-matrix
  float* attn      = (float*)(ws + (size_t)M_ * C_ * 2);     // 256 KB
  float* attn_norm = attn + M_;                              // 256 KB
  float* partial   = attn_norm + M_;                         // 128 KB

  pool_transpose_kernel<<<dim3(W_ / 64, C_ / 16, B_), 256, 0, stream>>>(x, poolT);
  attn_gemm_kernel<<<M_ / 128, 256, 0, stream>>>(poolT, Wg, proj, bias, attn);
  softmax_kernel<<<B_, 256, 0, stream>>>(attn, attn_norm);
  wsum_partial_kernel<<<dim3(16, B_), 128, 0, stream>>>(poolT, attn_norm, partial);
  wsum_reduce_kernel<<<(B_ * C_) / 256, 256, 0, stream>>>(partial, out);
}
